// ScoreAttention11_8778913153181
// MI455X (gfx1250) — matrix operand order repaired and checked by reading, not compiled, not run on hardware
//
#include <hip/hip_runtime.h>
#include <hip/hip_bf16.h>

// ---------------------------------------------------------------------------
// Sizes (fixed by the reference)
// ---------------------------------------------------------------------------
#define BATCH 16
#define SLEN  4096
#define DK    1024   // D_ATT == D_Q == K
#define DA    1024   // D_A   == N
#define NEG_BIG (-1.0e8f)

// GEMM tiling
#define BM 128
#define BN 128
#define BK 32
#define LDSTRIDE 40          // bf16 elements per LDS row (32 used + 8 pad -> conflict-light)

typedef __attribute__((ext_vector_type(16))) __bf16 v16bf;
typedef __attribute__((ext_vector_type(8)))  __bf16 v8bf;
typedef __attribute__((ext_vector_type(8)))  float  v8f;
typedef __attribute__((ext_vector_type(8)))  float  v8fl;

// ---------------------------------------------------------------------------
// Helpers
// ---------------------------------------------------------------------------
static __device__ __forceinline__ float fast_tanh(float x) {
#if __has_builtin(__builtin_amdgcn_tanhf)
  return __builtin_amdgcn_tanhf(x);          // v_tanh_f32 (TRANS32)
#elif __has_builtin(__builtin_amdgcn_tanh_f32)
  return __builtin_amdgcn_tanh_f32(x);
#else
  return tanhf(x);
#endif
}

// 8x f32 -> 8x bf16 (RNE) via native packed converts
static __device__ __forceinline__ uint4 cvt8_bf16(const float4 a, const float4 b) {
  v8fl f = {a.x, a.y, a.z, a.w, b.x, b.y, b.z, b.w};
  v8bf h = __builtin_convertvector(f, v8bf); // -> v_cvt_pk_bf16_f32 pairs
  union { v8bf v; uint4 q; } u;
  u.v = h;
  return u.q;
}

// Stage one 128x32 f32 tile (row stride 1024) -> LDS bf16 tile [128][LDSTRIDE].
// 256 threads: 2 threads per row, 16 elements each.
static __device__ __forceinline__ void stage_tile(unsigned short* __restrict__ dst,
                                                  const float* __restrict__ src,
                                                  int t, bool pf) {
  const int row  = t >> 1;
  const int half = (t & 1) << 4;
  const float* s = src + (size_t)row * DK + half;
  float4 f0 = *(const float4*)(s);
  float4 f1 = *(const float4*)(s + 4);
  float4 f2 = *(const float4*)(s + 8);
  float4 f3 = *(const float4*)(s + 12);
  if (pf) __builtin_prefetch(s + 2 * BK, 0, 1);   // -> global_prefetch_b8, 2 K-steps ahead
  uint4 q0 = cvt8_bf16(f0, f1);
  uint4 q1 = cvt8_bf16(f2, f3);
  unsigned short* d = dst + row * LDSTRIDE + half;       // byte addr = row*80 + {0,32}: 16B aligned
  *(uint4*)(d)     = q0;
  *(uint4*)(d + 8) = q1;
}

// ---------------------------------------------------------------------------
// Kernel 1: target[b,a] = x[b,:] . W_q[a,:] + b_q[a]      (16 x 1024, trivial)
// ---------------------------------------------------------------------------
__global__ void target_kernel(const float* __restrict__ x, const float* __restrict__ Wq,
                              const float* __restrict__ bq, float* __restrict__ tgt) {
  const int idx = blockIdx.x * blockDim.x + threadIdx.x;  // 16384 threads
  const int b = idx >> 10, a = idx & 1023;
  const float* xr = x + (b << 10);
  const float* wr = Wq + (size_t)a * DK;
  float s = bq[a];
  for (int d = 0; d < DK; d += 4) {
    float4 xv = *(const float4*)(xr + d);
    float4 wv = *(const float4*)(wr + d);
    s += xv.x * wv.x + xv.y * wv.y + xv.z * wv.z + xv.w * wv.w;
  }
  tgt[idx] = s;
}

// ---------------------------------------------------------------------------
// Kernel 2: main fused GEMM
//   precompute[m,n] = sum_k ctx[m,k]*Wpre[n,k] + bpre[n]      (written to d_out)
//   eacc[m]        += sum_n tanh(precompute[m,n]+target[b,n])*wv[n]
// Block = 128x128 tile, 8 waves of 32x64 (2x4 wmma accumulators), BK=32,
// double-buffered LDS, v_wmma_f32_16x16x32_bf16.
// ---------------------------------------------------------------------------
__global__ __launch_bounds__(256) void att_gemm_kernel(
    const float* __restrict__ ctx, const float* __restrict__ Wpre,
    const float* __restrict__ bpre, const float* __restrict__ target,
    const float* __restrict__ wv,
    float* __restrict__ pre_out, float* __restrict__ eacc) {
  __shared__ alignas(16) unsigned short As[2][BM * LDSTRIDE];
  __shared__ alignas(16) unsigned short Bs[2][BN * LDSTRIDE];

  const int t    = threadIdx.x;
  const int lane = t & 31;
  const int w    = t >> 5;
  const int wm   = w >> 1;        // 0..3 : wave M-tile (32 rows each)
  const int wn   = w & 1;         // 0..1 : wave N-tile (64 cols each)
  const int l15  = lane & 15;
  const int hsel = lane >> 4;     // 0/1 : half-wave select

  const int mBlk = blockIdx.y * BM;
  const int nBlk = blockIdx.x * BN;
  const float* actx = ctx  + (size_t)mBlk * DK;
  const float* bW   = Wpre + (size_t)nBlk * DK;

  stage_tile(As[0], actx, t, true);
  stage_tile(Bs[0], bW,   t, true);
  __syncthreads();

  const v8f vzero = {0.f, 0.f, 0.f, 0.f, 0.f, 0.f, 0.f, 0.f};
  v8f acc[2][4];
#pragma unroll
  for (int i = 0; i < 2; ++i)
#pragma unroll
    for (int j = 0; j < 4; ++j) acc[i][j] = vzero;

  int buf = 0;
  const int KSTEPS = DK / BK;     // 32
  for (int kt = 0; kt < KSTEPS; ++kt) {
    if (kt + 1 < KSTEPS) {
      stage_tile(As[buf ^ 1], actx + (size_t)(kt + 1) * BK, t, kt + 3 < KSTEPS);
      stage_tile(Bs[buf ^ 1], bW   + (size_t)(kt + 1) * BK, t, kt + 3 < KSTEPS);
    }

    // A fragments: lanes 0-15 hold K 0..15, lanes 16-31 hold K 16..31 (contiguous).
    v16bf afr[2], bfr[4];
    const int kbA = hsel << 4;
#pragma unroll
    for (int mf = 0; mf < 2; ++mf) {
      const unsigned short* p = &As[buf][(wm * 32 + mf * 16 + l15) * LDSTRIDE + kbA];
      union { uint4 q[2]; v16bf v; } u;
      u.q[0] = *(const uint4*)(p);
      u.q[1] = *(const uint4*)(p + 8);    // 16 contiguous K values
      afr[mf] = u.v;
    }
    // B fragments: 32x16 B, lane = column n; lanes 0-15 hold K 0..15, lanes 16-31 hold K 16..31.
    const int kbB = hsel << 4;
#pragma unroll
    for (int nf = 0; nf < 4; ++nf) {
      const unsigned short* p = &Bs[buf][(wn * 64 + nf * 16 + l15) * LDSTRIDE + kbB];
      union { uint4 q[2]; v16bf v; } u;
      u.q[0] = *(const uint4*)(p);
      u.q[1] = *(const uint4*)(p + 8);    // 16 contiguous K values
      bfr[nf] = u.v;
    }

#pragma unroll
    for (int mf = 0; mf < 2; ++mf)
#pragma unroll
      for (int nf = 0; nf < 4; ++nf)
        acc[mf][nf] = __builtin_amdgcn_wmma_f32_16x16x32_bf16(
            false, afr[mf], false, bfr[nf], (short)0, acc[mf][nf], false, false);

    __syncthreads();
    buf ^= 1;
  }

  // ------------------------- fused epilogue -------------------------
  // C/D layout: VGPR i of lane L holds (M = i + 8*(L>=16), N = L&15) of the 16x16 tile.
  const int bIdx = mBlk >> 12;              // 4096 % 128 == 0 -> uniform per block
  const float* tgt = target + (bIdx << 10);
  float ep[2][8] = {};

#pragma unroll
  for (int mf = 0; mf < 2; ++mf)
#pragma unroll
    for (int nf = 0; nf < 4; ++nf) {
      const int n   = nBlk + wn * 64 + nf * 16 + l15;
      const float bp = bpre[n];
      const float wvn = wv[n];
      const float tg  = tgt[n];
#pragma unroll
      for (int i = 0; i < 8; ++i) {
        const int mloc = wm * 32 + mf * 16 + (hsel << 3) + i;
        const size_t mg = (size_t)(mBlk + mloc);
        const float val = acc[mf][nf][i] + bp;
        pre_out[mg * DA + n] = val;
        ep[mf][i] += fast_tanh(val + tg) * wvn;
      }
    }

  // Reduce across the 16 lanes that share one M row (xor 8,4,2,1 stays in each half).
#pragma unroll
  for (int mf = 0; mf < 2; ++mf)
#pragma unroll
    for (int i = 0; i < 8; ++i) {
      float v = ep[mf][i];
      v += __shfl_xor(v, 8, 32);
      v += __shfl_xor(v, 4, 32);
      v += __shfl_xor(v, 2, 32);
      v += __shfl_xor(v, 1, 32);
      if (l15 == 0) {
        const int mloc = wm * 32 + mf * 16 + (hsel << 3) + i;
        atomicAdd(&eacc[mBlk + mloc], v);
      }
    }
}

// ---------------------------------------------------------------------------
// Kernel 3: masked softmax over S per batch. 16 blocks x 256 threads.
// ---------------------------------------------------------------------------
static __device__ __forceinline__ float block_reduce(float v, bool domax, float* red, int t) {
  const int lane = t & 31, w = t >> 5;
#pragma unroll
  for (int off = 16; off; off >>= 1) {
    const float o = __shfl_xor(v, off, 32);
    v = domax ? fmaxf(v, o) : (v + o);
  }
  if (lane == 0) red[w] = v;
  __syncthreads();
  float r = red[0];
#pragma unroll
  for (int i = 1; i < 8; ++i) r = domax ? fmaxf(r, red[i]) : (r + red[i]);
  __syncthreads();
  return r;
}

__global__ void softmax_kernel(const float* __restrict__ eacc, const int* __restrict__ mask,
                               const float* __restrict__ bv, float* __restrict__ eout) {
  __shared__ float red[8];
  const int b = blockIdx.x, t = threadIdx.x;
  const float* ein = eacc + (b << 12);
  const int*   mk  = mask + (b << 12);
  const float bias = bv[0];

  float v[16];
  float lmax = -3.4e38f;
#pragma unroll
  for (int j = 0; j < 16; ++j) {
    const int s = t + (j << 8);
    const float m = (float)mk[s];
    const float e = (ein[s] + bias) * (1.0f - m) + m * NEG_BIG;
    v[j] = e;
    lmax = fmaxf(lmax, e);
  }
  const float gmax = block_reduce(lmax, true, red, t);

  float lsum = 0.f;
#pragma unroll
  for (int j = 0; j < 16; ++j) { v[j] = __expf(v[j] - gmax); lsum += v[j]; }
  const float gsum = block_reduce(lsum, false, red, t);

  const float inv = 1.0f / gsum;
  float* o = eout + (b << 12);
#pragma unroll
  for (int j = 0; j < 16; ++j) o[t + (j << 8)] = v[j] * inv;
}

// ---------------------------------------------------------------------------
// Launch
// ---------------------------------------------------------------------------
extern "C" void kernel_launch(void* const* d_in, const int* in_sizes, int n_in,
                              void* d_out, int out_size, void* d_ws, size_t ws_size,
                              hipStream_t stream) {
  (void)in_sizes; (void)n_in; (void)out_size; (void)ws_size;
  const float* x    = (const float*)d_in[0];
  const float* ctx  = (const float*)d_in[1];
  const int*   mask = (const int*)d_in[2];
  const float* Wpre = (const float*)d_in[3];
  const float* bpre = (const float*)d_in[4];
  const float* Wq   = (const float*)d_in[5];
  const float* bq   = (const float*)d_in[6];
  const float* wv   = (const float*)d_in[7];
  const float* bv   = (const float*)d_in[8];

  // d_out = [energy (B*S)] ++ [precompute (B*S*DA)]
  float* out_energy = (float*)d_out;
  float* out_pre    = out_energy + BATCH * SLEN;

  // workspace: target (16*1024 f32), energy accumulator (B*S f32, must be zeroed)
  float* tgt  = (float*)d_ws;
  float* eacc = tgt + BATCH * DA;

  hipMemsetAsync(eacc, 0, (size_t)BATCH * SLEN * sizeof(float), stream);

  target_kernel<<<(BATCH * DA) / 256, 256, 0, stream>>>(x, Wq, bq, tgt);

  dim3 grid(DA / BN, (BATCH * SLEN) / BM);  // (8, 512)
  att_gemm_kernel<<<grid, 256, 0, stream>>>(ctx, Wpre, bpre, tgt, wv, out_pre, eacc);

  softmax_kernel<<<BATCH, 256, 0, stream>>>(eacc, mask, bv, out_energy);
}
